// sinc_conv_446676598825
// MI455X (gfx1250) — matrix operand order repaired and checked by reading, not compiled, not run on hardware
//
#include <hip/hip_runtime.h>
#include <hip/hip_bf16.h>
#include <math.h>

typedef __attribute__((ext_vector_type(16))) _Float16 v16h;
typedef __attribute__((ext_vector_type(8)))  float    v8f;

#define SC_NFILT   80
#define SC_FDIM    251
#define SC_KPAD    256      // 251 padded to 8 chunks of K=32
#define SC_LIN     32000
#define SC_LOUT    31750    // 32000 - 251 + 1
#define SC_BATCH   32
#define SC_MTILES  5        // 80 filters / 16
#define SC_PI      3.14159265358979323846f

// ---------------------------------------------------------------------------
// Kernel 1: build the 80 SincNet filters (fp32 math), normalize by row max,
// apply Hamming window, emit f16 K-padded [80][256] matrix into workspace.
// One block (256 threads) per filter. Runtime-negligible (80 blocks).
// ---------------------------------------------------------------------------
__global__ void __launch_bounds__(256)
sc_build_filters(const float* __restrict__ b1,
                 const float* __restrict__ band,
                 _Float16* __restrict__ filt16) {
    const int f = blockIdx.x;      // filter index 0..79
    const int j = threadIdx.x;     // tap index 0..255 (251 real taps)
    __shared__ float red[256];

    const float inv_fs = 1.0f / 16000.0f;
    const float fb = fabsf(b1[f])   + 50.0f * inv_fs;              // filt_beg
    const float fe = fb + fabsf(band[f]) + 50.0f * inv_fs;         // filt_end

    float bp = 0.0f;
    if (j < SC_FDIM) {
        if (j == 125) {
            bp = 2.0f * fe - 2.0f * fb;                            // sinc(0)=1
        } else {
            // t = |j-125|/fs ; arg = 2*pi*(f_norm*fs)*t
            const float t  = fabsf((float)(j - 125)) * inv_fs;
            const float a1 = 2.0f * SC_PI * (fb * 16000.0f) * t;
            const float a2 = 2.0f * SC_PI * (fe * 16000.0f) * t;
            bp = 2.0f * fe * (sinf(a2) / a2) - 2.0f * fb * (sinf(a1) / a1);
        }
    }

    // row-max reduction (max is positive: center tap = 2*(fe-fb) > 0)
    red[j] = (j < SC_FDIM) ? bp : -3.4e38f;
    __syncthreads();
    #pragma unroll
    for (int s = 128; s > 0; s >>= 1) {
        if (j < s) red[j] = fmaxf(red[j], red[j + s]);
        __syncthreads();
    }
    const float mx = red[0];

    // Hamming window: n[j] = j*251/250 -> w = 0.54 - 0.46*cos(2*pi*j/250)
    const float w = 0.54f - 0.46f * cosf(2.0f * SC_PI * (float)j * (1.0f / 250.0f));
    const float val = (j < SC_FDIM) ? (bp / mx) * w : 0.0f;  // zero K-padding
    filt16[f * SC_KPAD + j] = (_Float16)val;
}

// ---------------------------------------------------------------------------
// Kernel 2: implicit-GEMM conv via v_wmma_f32_16x16x32_f16.
//   M = filter (all 5 16-tiles per block), N = output time (16 per wave,
//   128 per block), K = taps padded to 256 = 8 x 32.
// Each wave builds one B fragment per K-chunk and reuses it across all 5
// filter tiles -> 40 WMMAs per wave per block.
// Output is 325 MB (> 192 MB L2): stream results with NT stores so the
// x windows (250-sample halo re-read by neighbor blocks) and the 40 KB
// filter matrix stay cache-resident.
// Block = 256 threads = 8 wave32; grid = (ceil(31750/128), 32).
// ---------------------------------------------------------------------------
__global__ void __launch_bounds__(256)
sc_conv_wmma(const float* __restrict__ x,
             const _Float16* __restrict__ filt16,
             float* __restrict__ out) {
    // x window for this block: 128 outputs + 250 halo = 378 samples, pad 384
    __shared__ _Float16 xlds[384];

    const int tid  = threadIdx.x;
    const int lane = tid & 31;
    const int wave = tid >> 5;
    const int t0   = blockIdx.x * 128;    // first output position of block
    const int b    = blockIdx.y;          // batch

    const float* __restrict__ xb = x + (size_t)b * SC_LIN;

    // stage + downconvert x window into LDS (zero-fill halo/OOB)
    for (int i = tid; i < 384; i += 256) {
        const int g = t0 + i;
        const float v = (i < 378 && g < SC_LIN) ? xb[g] : 0.0f;
        xlds[i] = (_Float16)v;
    }
    __syncthreads();

    const int col   = lane & 15;   // N column (time) / A-matrix row select
    const int hi    = lane >> 4;   // half-wave select
    const int akoff = hi * 16;     // A: lanes 16-31 hold K 16..31
    const int bkoff = hi * 16;     // B: lanes 16-31 hold K 16..31
    const int tl    = wave * 16;   // wave's local time base

    v8f acc[SC_MTILES];
    #pragma unroll
    for (int mt = 0; mt < SC_MTILES; ++mt) acc[mt] = (v8f){};

    #pragma unroll
    for (int c = 0; c < 8; ++c) {
        // ---- B fragment: signal, 16 consecutive K values per lane (LDS) ----
        const _Float16* __restrict__ xs = &xlds[tl + col + c * 32 + bkoff];
        v16h bf;
        #pragma unroll
        for (int s = 0; s < 16; ++s) bf[s] = xs[s];

        // ---- reuse B across all 5 filter tiles ----
        #pragma unroll
        for (int mt = 0; mt < SC_MTILES; ++mt) {
            // A fragment: filters, 16-bit A layout (16x32 per wave), L1-hot
            const _Float16* __restrict__ fw =
                filt16 + (mt * 16 + col) * SC_KPAD + c * 32 + akoff;
            v16h a;
            #pragma unroll
            for (int v = 0; v < 4; ++v) {
                a[2 * v]         = fw[2 * v];          // K pair (2v, 2v+1)
                a[2 * v + 1]     = fw[2 * v + 1];
                a[8 + 2 * v]     = fw[8 + 2 * v];      // K pair (+8)
                a[8 + 2 * v + 1] = fw[8 + 2 * v + 1];
            }
            // D = A x B + C  (f32 accumulate)
            acc[mt] = __builtin_amdgcn_wmma_f32_16x16x32_f16(
                /*neg_a=*/false, a, /*neg_b=*/false, bf,
                /*c_mod=*/(short)0, acc[mt],
                /*reuse_a=*/false, /*reuse_b=*/false);
        }
    }

    // C layout: VGPR r -> rows (r, r+8); lanes 0-15 N=0..15, 16-31 N=0..15
    // Stream results past the cache hierarchy (write-once, never re-read).
    const int t = t0 + tl + col;
    if (t < SC_LOUT) {
        #pragma unroll
        for (int mt = 0; mt < SC_MTILES; ++mt) {
            #pragma unroll
            for (int r = 0; r < 8; ++r) {
                const int fr = mt * 16 + r + hi * 8;
                float* dst =
                    &out[(size_t)b * SC_NFILT * SC_LOUT + (size_t)fr * SC_LOUT + t];
#if __has_builtin(__builtin_nontemporal_store)
                __builtin_nontemporal_store(acc[mt][r], dst);
#else
                *dst = acc[mt][r];
#endif
            }
        }
    }
}

// ---------------------------------------------------------------------------
extern "C" void kernel_launch(void* const* d_in, const int* in_sizes, int n_in,
                              void* d_out, int out_size, void* d_ws, size_t ws_size,
                              hipStream_t stream) {
    const float* x     = (const float*)d_in[0];   // (32,1,32000)
    const float* b1    = (const float*)d_in[1];   // (80,)
    const float* band  = (const float*)d_in[2];   // (80,)
    float*       out   = (float*)d_out;           // (32,80,31750)
    _Float16*    f16ws = (_Float16*)d_ws;         // 80*256 halfs = 40 KB

    // Phase 1: build filters (80 blocks x 256 threads)
    sc_build_filters<<<dim3(SC_NFILT), dim3(256), 0, stream>>>(b1, band, f16ws);

    // Phase 2: WMMA conv. 128 outputs/block -> ceil(31750/128) = 249 tiles
    const int ntiles = (SC_LOUT + 127) / 128;
    sc_conv_wmma<<<dim3(ntiles, SC_BATCH), dim3(256), 0, stream>>>(x, f16ws, out);
}
